// RANetBlockB_91242285236794
// MI455X (gfx1250) — 8-bit matrix operand order repaired and checked by reading, compiled, not run on hardware
//
#include <hip/hip_runtime.h>

typedef __attribute__((ext_vector_type(8))) int v8i;

#define NB 32
#define CH 256
#define HH 56
#define WW 56
#define HP 58
#define EPS 1e-5f

__device__ inline v8i load_frag64(const unsigned char* p) {
    union { v8i v; unsigned long long q[4]; } u;
    u.q[0] = *(const unsigned long long*)(p);
    u.q[1] = *(const unsigned long long*)(p + 8);
    u.q[2] = *(const unsigned long long*)(p + 32);
    u.q[3] = *(const unsigned long long*)(p + 40);
    return u.v;
}
__device__ inline v8i load_fragB(const unsigned char* p) {
    union { v8i v; int4 u[2]; } u;
    u.u[0] = *(const int4*)(p);
    u.u[1] = *(const int4*)(p + 16);
    return u.v;
}

__device__ inline v8i wmma_iu8(v8i a, v8i b, v8i c) {
    return __builtin_amdgcn_wmma_i32_16x16x64_iu8(true, a, true, b, c, false, false);
}

__global__ __launch_bounds__(256) void k_chan_consts(
    const float* __restrict__ w3, const float* __restrict__ g1, const float* __restrict__ be1,
    const float* __restrict__ mn1, const float* __restrict__ vr1,
    const float* __restrict__ wpw, const float* __restrict__ g2, const float* __restrict__ be2,
    const float* __restrict__ mn2, const float* __restrict__ vr2,
    float* __restrict__ cc)
{
    int oc = threadIdx.x;
    float s = 0.f;
    const float* p = w3 + (size_t)oc * (CH * 9);
    for (int i = 0; i < CH * 9; ++i) s += fabsf(p[i]);
    s *= (1.f / (CH * 9));
    float inv1 = g1[oc] * rsqrtf(vr1[oc] + EPS);
    cc[oc]        = s * inv1;
    cc[256 + oc]  = be1[oc] - mn1[oc] * inv1;

    float s2 = 0.f;
    const float* q = wpw + (size_t)oc * CH;
    for (int i = 0; i < CH; ++i) s2 += fabsf(q[i]);
    s2 *= (1.f / CH);
    float inv2 = g2[oc] * rsqrtf(vr2[oc] + EPS);
    cc[512 + oc] = s2 * inv2;
    cc[768 + oc] = be2[oc] - mn2[oc] * inv2;
}

__global__ __launch_bounds__(256) void k_pack_w3(const float* __restrict__ w3,
                                                 unsigned int* __restrict__ pb)
{
    int t  = blockIdx.x * 256 + threadIdx.x;
    int j  = t & 7;
    int L  = (t >> 3) & 31;
    int ot = (t >> 8) & 15;
    int kc = t >> 12;
    int base = (j < 4) ? ((L < 16) ? 0 : 16) : ((L < 16) ? 32 : 48);
    int oc = ot * 16 + (L & 15);
    unsigned int word = 0;
    #pragma unroll
    for (int b = 0; b < 4; ++b) {
        int K   = kc * 64 + base + ((j & 3) << 2) + b;
        int tap = K >> 8;
        int ic  = K & 255;
        float w = w3[((size_t)oc * CH + ic) * 9 + tap];
        int s   = (w > 0.f) - (w < 0.f);
        word |= ((unsigned int)(unsigned char)(char)s) << (8 * b);
    }
    pb[t] = word;
}

__global__ __launch_bounds__(256) void k_pack_pw(const float* __restrict__ wpw,
                                                 unsigned int* __restrict__ pb)
{
    int t  = blockIdx.x * 256 + threadIdx.x;
    int j  = t & 7;
    int L  = (t >> 3) & 31;
    int ot = (t >> 8) & 15;
    int kc = t >> 12;
    int base = (j < 4) ? ((L < 16) ? 0 : 16) : ((L < 16) ? 32 : 48);
    int oc = ot * 16 + (L & 15);
    unsigned int word = 0;
    #pragma unroll
    for (int b = 0; b < 4; ++b) {
        int ic  = kc * 64 + base + ((j & 3) << 2) + b;
        float w = wpw[(size_t)oc * CH + ic];
        int s   = (w > 0.f) - (w < 0.f);
        word |= ((unsigned int)(unsigned char)(char)s) << (8 * b);
    }
    pb[t] = word;
}

__global__ __launch_bounds__(256) void k_binarize(const float* __restrict__ x,
                                                  const float* __restrict__ mv1,
                                                  unsigned int* __restrict__ sact)
{
    int t  = blockIdx.x * 256 + threadIdx.x;
    int c4 = t & 63;
    int wp = (t >> 6) % HP;
    int hp = (t / (64 * HP)) % HP;
    int n  = t / (64 * HP * HP);
    unsigned int word = 0;
    if (hp >= 1 && hp <= HH && wp >= 1 && wp <= WW) {
        int h = hp - 1, w = wp - 1, c = c4 * 4;
        #pragma unroll
        for (int k = 0; k < 4; ++k) {
            float v = x[(((size_t)n * CH + (c + k)) * HH + h) * WW + w] + mv1[c + k];
            int s = (v > 0.f) - (v < 0.f);
            word |= ((unsigned int)(unsigned char)(char)s) << (8 * k);
        }
    }
    sact[t] = word;
}

__global__ __launch_bounds__(256) void k_fused(
    const unsigned char* __restrict__ sact,
    const unsigned char* __restrict__ pb3,
    const unsigned char* __restrict__ pbpw,
    const float* __restrict__ cc,
    const float* __restrict__ x,
    const float* __restrict__ p1m, const float* __restrict__ p1s, const float* __restrict__ p1b,
    const float* __restrict__ mv2,
    const float* __restrict__ p2m, const float* __restrict__ p2s, const float* __restrict__ p2b,
    float* __restrict__ out)
{
    __shared__ float out1_lds[64 * CH];
    __shared__ __align__(16) unsigned char act2_lds[64 * 272];

    int bid = blockIdx.x;
    int wt  = bid & 3;
    int ht  = (bid >> 2) % 14;
    int n   = bid / 56;
    int w0  = wt * 16;
    int h0  = ht * 4;

    int tid  = threadIdx.x;
    int wave = tid >> 5;
    int L    = tid & 31;
    int col  = L & 15;
    int half = L >> 4;
    int off8 = half ? 16 : 0;
    int ot0  = wave * 2, ot1 = wave * 2 + 1;

    int mwc = w0 + col; if (mwc > WW - 1) mwc = WW - 1;

    const unsigned char* abase0 = sact + (((size_t)n * HP + (h0 + 0)) * HP + mwc) * CH + off8;
    const unsigned char* abase1 = sact + (((size_t)n * HP + (h0 + 1)) * HP + mwc) * CH + off8;
    const unsigned char* abase2 = sact + (((size_t)n * HP + (h0 + 2)) * HP + mwc) * CH + off8;
    const unsigned char* abase3 = sact + (((size_t)n * HP + (h0 + 3)) * HP + mwc) * CH + off8;
    const unsigned char* bbase0 = pb3 + (ot0 * 32 + L) * 32;
    const unsigned char* bbase1 = pb3 + (ot1 * 32 + L) * 32;

    v8i acc[4][2];
    #pragma unroll
    for (int mt = 0; mt < 4; ++mt) { acc[mt][0] = {}; acc[mt][1] = {}; }

    #pragma unroll
    for (int tap = 0; tap < 9; ++tap) {
        const int toff = ((tap / 3) * HP + (tap % 3)) * CH;
        #pragma unroll
        for (int icc = 0; icc < 4; ++icc) {
            const int kc   = tap * 4 + icc;
            const int aoff = toff + icc * 64;
            v8i b0 = load_fragB(bbase0 + kc * 16384);
            v8i b1 = load_fragB(bbase1 + kc * 16384);
            v8i a0 = load_frag64(abase0 + aoff);
            v8i a1 = load_frag64(abase1 + aoff);
            v8i a2 = load_frag64(abase2 + aoff);
            v8i a3 = load_frag64(abase3 + aoff);
            acc[0][0] = wmma_iu8(a0, b0, acc[0][0]);
            acc[0][1] = wmma_iu8(a0, b1, acc[0][1]);
            acc[1][0] = wmma_iu8(a1, b0, acc[1][0]);
            acc[1][1] = wmma_iu8(a1, b1, acc[1][1]);
            acc[2][0] = wmma_iu8(a2, b0, acc[2][0]);
            acc[2][1] = wmma_iu8(a2, b1, acc[2][1]);
            acc[3][0] = wmma_iu8(a3, b0, acc[3][0]);
            acc[3][1] = wmma_iu8(a3, b1, acc[3][1]);
        }
    }

    const float* a1c = cc;       const float* b1c = cc + 256;
    #pragma unroll
    for (int t = 0; t < 2; ++t) {
        int oc = (wave * 2 + t) * 16 + col;
        float A1 = a1c[oc], B1 = b1c[oc];
        float M1 = p1m[oc], S1 = p1s[oc], Bs1 = p1b[oc], Mv2 = mv2[oc];
        #pragma unroll
        for (int mt = 0; mt < 4; ++mt) {
            int h = h0 + mt;
            const float* xrow = x + (((size_t)n * CH + oc) * HH + h) * WW;
            #pragma unroll
            for (int g = 0; g < 8; ++g) {
                int m = half * 8 + g;
                int wc = w0 + m; if (wc > WW - 1) wc = WW - 1;
                float v  = (float)acc[mt][t][g] * A1 + B1 + xrow[wc];
                float tt = v + M1; tt = tt > 0.f ? tt : S1 * tt;
                float o1 = tt + Bs1;
                int mi = mt * 16 + m;
                out1_lds[mi * CH + oc] = o1;
                float sv = o1 + Mv2;
                act2_lds[mi * 272 + oc] = (unsigned char)(char)((sv > 0.f) - (sv < 0.f));
            }
        }
    }
    __syncthreads();

    const unsigned char* a2base  = act2_lds + col * 272 + off8;
    const unsigned char* b2base0 = pbpw + (ot0 * 32 + L) * 32;
    const unsigned char* b2base1 = pbpw + (ot1 * 32 + L) * 32;

    v8i c2[4][2];
    #pragma unroll
    for (int mt = 0; mt < 4; ++mt) { c2[mt][0] = {}; c2[mt][1] = {}; }
    #pragma unroll
    for (int kc = 0; kc < 4; ++kc) {
        v8i b0 = load_fragB(b2base0 + kc * 16384);
        v8i b1 = load_fragB(b2base1 + kc * 16384);
        #pragma unroll
        for (int mt = 0; mt < 4; ++mt) {
            v8i a = load_frag64(a2base + mt * (16 * 272) + kc * 64);
            c2[mt][0] = wmma_iu8(a, b0, c2[mt][0]);
            c2[mt][1] = wmma_iu8(a, b1, c2[mt][1]);
        }
    }

    const float* a2c = cc + 512; const float* b2c = cc + 768;
    #pragma unroll
    for (int t = 0; t < 2; ++t) {
        int oc = (wave * 2 + t) * 16 + col;
        float A2 = a2c[oc], B2 = b2c[oc];
        float M2 = p2m[oc], S2 = p2s[oc], Bs2 = p2b[oc];
        #pragma unroll
        for (int mt = 0; mt < 4; ++mt) {
            int h = h0 + mt;
            float* orow = out + (((size_t)n * CH + oc) * HH + h) * WW;
            #pragma unroll
            for (int g = 0; g < 8; ++g) {
                int m = half * 8 + g;
                int w = w0 + m;
                int mi = mt * 16 + m;
                float v  = (float)c2[mt][t][g] * A2 + B2 + out1_lds[mi * CH + oc];
                float tt = v + M2; tt = tt > 0.f ? tt : S2 * tt;
                float o2 = tt + Bs2;
                if (w < WW) orow[w] = o2;
            }
        }
    }
}

extern "C" void kernel_launch(void* const* d_in, const int* in_sizes, int n_in,
                              void* d_out, int out_size, void* d_ws, size_t ws_size,
                              hipStream_t stream) {
    (void)in_sizes; (void)n_in; (void)out_size; (void)ws_size;
    const float* x   = (const float*)d_in[0];
    const float* mv1 = (const float*)d_in[1];
    const float* w3  = (const float*)d_in[2];
    const float* g1  = (const float*)d_in[3];
    const float* be1 = (const float*)d_in[4];
    const float* mn1 = (const float*)d_in[5];
    const float* vr1 = (const float*)d_in[6];
    const float* p1m = (const float*)d_in[7];
    const float* p1s = (const float*)d_in[8];
    const float* p1b = (const float*)d_in[9];
    const float* mv2 = (const float*)d_in[10];
    const float* wpw = (const float*)d_in[11];
    const float* g2  = (const float*)d_in[12];
    const float* be2 = (const float*)d_in[13];
    const float* mn2 = (const float*)d_in[14];
    const float* vr2 = (const float*)d_in[15];
    const float* p2m = (const float*)d_in[16];
    const float* p2s = (const float*)d_in[17];
    const float* p2b = (const float*)d_in[18];
    float* out = (float*)d_out;

    char* ws = (char*)d_ws;
    size_t off_sact = 0;
    size_t sz_sact  = (size_t)NB * HP * HP * CH;
    size_t off_pb3  = off_sact + sz_sact;
    size_t sz_pb3   = (size_t)36 * 16 * 32 * 32;
    size_t off_pbpw = off_pb3 + sz_pb3;
    size_t sz_pbpw  = (size_t)4 * 16 * 32 * 32;
    size_t off_cc   = off_pbpw + sz_pbpw;

    unsigned char* sact = (unsigned char*)(ws + off_sact);
    unsigned int*  pb3  = (unsigned int*)(ws + off_pb3);
    unsigned int*  pbpw = (unsigned int*)(ws + off_pbpw);
    float*         cc   = (float*)(ws + off_cc);

    k_chan_consts<<<1, 256, 0, stream>>>(w3, g1, be1, mn1, vr1, wpw, g2, be2, mn2, vr2, cc);
    k_pack_w3<<<576, 256, 0, stream>>>(w3, pb3);
    k_pack_pw<<<64, 256, 0, stream>>>(wpw, pbpw);
    k_binarize<<<(int)(sz_sact / 4 / 256), 256, 0, stream>>>(x, mv1, (unsigned int*)sact);
    k_fused<<<NB * 14 * 4, 256, 0, stream>>>(sact, (const unsigned char*)pb3,
                                             (const unsigned char*)pbpw, cc, x,
                                             p1m, p1s, p1b, mv2, p2m, p2s, p2b, out);
}
